// ConvZero_64295660421817
// MI455X (gfx1250) — matrix operand order repaired and checked by reading, not compiled, not run on hardware
//
#include <hip/hip_runtime.h>
#include <hip/hip_bf16.h>

typedef __attribute__((ext_vector_type(16))) __bf16 v16bf;
typedef __attribute__((ext_vector_type(8)))  float  v8f;

#define BM 64            // block rows
#define BN 128           // block cols
#define BK 32            // K step
#define LDS_STRIDE 40    // ushorts per row (32 data + 8 pad) => 80B rows, 16B-aligned chunks

__device__ __forceinline__ unsigned short f2bfbits(float f) {
  unsigned u = __float_as_uint(f);
  unsigned r = u + 0x7FFFu + ((u >> 16) & 1u);   // round-to-nearest-even
  return (unsigned short)(r >> 16);
}

union FragBF { uint4 u4[2]; v16bf v; };

// C[M,N] = act(A[M,K]) @ B[K,N]  with optional fused per-column BN+ReLU on A,
// optional bias on C, optional column sum/sumsq accumulation for the next BN.
// Block: 256 threads = 8 waves in a 2(row) x 4(col) grid; each wave owns a
// 32x32 macro-tile = four 16x16x32 bf16 WMMA accumulators.
__global__ __launch_bounds__(256) void gemm_bf16_wmma(
    const float* __restrict__ A, int M, int K,
    const float* __restrict__ B, int ldb,
    float* __restrict__ C, int N,
    const float* __restrict__ aScale, const float* __restrict__ aShift,
    const float* __restrict__ bias,
    float* __restrict__ colSum, float* __restrict__ colSumSq)
{
  __shared__ unsigned short ldsA[BM * LDS_STRIDE];   // 64 x 32 bf16 tile of A
  __shared__ unsigned short ldsB[BN * LDS_STRIDE];   // B^T: 128 x 32 bf16

  const int tid  = threadIdx.x;
  const int lane = tid & 31;
  const int w    = tid >> 5;       // wave 0..7
  const int wr   = w >> 2;         // wave row 0..1  (32-row slab)
  const int wc   = w & 3;          // wave col 0..3  (32-col slab)
  const int row0 = blockIdx.x * BM;
  const int n0   = blockIdx.y * BN;

  v8f acc[2][2];
  #pragma unroll
  for (int i = 0; i < 2; ++i)
    #pragma unroll
    for (int j = 0; j < 2; ++j) acc[i][j] = (v8f){};

  for (int k0 = 0; k0 < K; k0 += BK) {
    // ---- stage A tile (fp32 -> BN+ReLU -> bf16) into LDS: 64x32, 8 elts/thread ----
    {
      const int r   = tid >> 2;            // 0..63
      const int kb  = (tid & 3) * 8;       // 0,8,16,24
      const int row = row0 + r;
      float vals[8];
      if (row < M) {
        const float4* ap = (const float4*)(A + (size_t)row * K + k0 + kb);
        #pragma unroll
        for (int j = 0; j < 2; ++j) {
          float4 f = ap[j];
          vals[4*j+0] = f.x; vals[4*j+1] = f.y; vals[4*j+2] = f.z; vals[4*j+3] = f.w;
        }
        if (aScale) {
          #pragma unroll
          for (int j = 0; j < 8; ++j) {
            int kk = k0 + kb + j;
            vals[j] = fmaxf(vals[j] * aScale[kk] + aShift[kk], 0.0f);
          }
        }
        if (k0 + BK < K)                      // software prefetch next K slab
          __builtin_prefetch(A + (size_t)row * K + k0 + BK + kb, 0, 0);
      } else {
        #pragma unroll
        for (int j = 0; j < 8; ++j) vals[j] = 0.0f;   // zero-fill keeps stats exact
      }
      unsigned* dp = (unsigned*)&ldsA[r * LDS_STRIDE + kb];
      #pragma unroll
      for (int j = 0; j < 4; ++j)
        dp[j] = (unsigned)f2bfbits(vals[2*j]) | ((unsigned)f2bfbits(vals[2*j+1]) << 16);
    }
    // ---- stage B tile (32x128) transposed into LDS: 16 elts/thread ----
    {
      const int k  = tid >> 3;             // 0..31
      const int nb = (tid & 7) * 16;       // 0..112
      const float4* bp = (const float4*)(B + (size_t)(k0 + k) * ldb + n0 + nb);
      #pragma unroll
      for (int j4 = 0; j4 < 4; ++j4) {
        float4 f = bp[j4];
        const int n = nb + 4 * j4;
        ldsB[(n + 0) * LDS_STRIDE + k] = f2bfbits(f.x);
        ldsB[(n + 1) * LDS_STRIDE + k] = f2bfbits(f.y);
        ldsB[(n + 2) * LDS_STRIDE + k] = f2bfbits(f.z);
        ldsB[(n + 3) * LDS_STRIDE + k] = f2bfbits(f.w);
      }
    }
    __syncthreads();

    // ---- per-wave fragments: 2 A frags + 2 B frags -> 4 WMMAs ----
    FragBF fa[2], fb[2];
    #pragma unroll
    for (int f = 0; f < 2; ++f) {
      // A frag: lanes 0-15 hold M=lane, K {0..7,16..23}; lanes 16-31 M=lane-16, K {8..15,24..31}
      const int m  = wr * 32 + f * 16 + (lane & 15);
      const int ko = (lane & 16) ? 8 : 0;
      const unsigned short* ap = &ldsA[m * LDS_STRIDE];
      fa[f].u4[0] = *(const uint4*)(ap + ko);
      fa[f].u4[1] = *(const uint4*)(ap + ko + 16);
      // B frag: lanes 0-15 col N=lane K {0..7,16..23}; lanes 16-31 col N=lane-16 K {8..15,24..31}
      const int n  = wc * 32 + f * 16 + (lane & 15);
      const int kb = (lane & 16) ? 8 : 0;
      const unsigned short* bp = &ldsB[n * LDS_STRIDE + kb];
      fb[f].u4[0] = *(const uint4*)(bp);
      fb[f].u4[1] = *(const uint4*)(bp + 16);
    }
    #pragma unroll
    for (int i = 0; i < 2; ++i)
      #pragma unroll
      for (int j = 0; j < 2; ++j)
        acc[i][j] = __builtin_amdgcn_wmma_f32_16x16x32_bf16(false, fa[i].v, false, fb[j].v,
                                                            (short)0, acc[i][j], false, false);
    __syncthreads();
  }

  // ---- epilogue: store + optional bias + BN-stat side-band ----
  #pragma unroll
  for (int j = 0; j < 2; ++j) {
    const int col = n0 + wc * 32 + j * 16 + (lane & 15);
    const float bs = bias ? bias[col] : 0.0f;
    float s = 0.f, q = 0.f;
    #pragma unroll
    for (int i = 0; i < 2; ++i) {
      const int mb = row0 + wr * 32 + i * 16 + ((lane & 16) ? 8 : 0);
      #pragma unroll
      for (int r = 0; r < 8; ++r) {
        int m = mb + r;
        float v = acc[i][j][r] + bs;
        if (m < M) C[(size_t)m * N + col] = v;
        s += v; q += v * v;
      }
    }
    if (colSum) { atomicAdd(&colSum[col], s); atomicAdd(&colSumSq[col], q); }
  }
}

// messages[e] = P1[src[e]] + P2[dst[e]] + P3[face[e]]  (+ column sum/sumsq for BN1a)
__global__ __launch_bounds__(256) void msg_kernel(
    const float* __restrict__ P1, const float* __restrict__ P2, const float* __restrict__ P3,
    const int* __restrict__ src, const int* __restrict__ dst, const int* __restrict__ face,
    float* __restrict__ msg, float* __restrict__ colSum, float* __restrict__ colSumSq, int E)
{
  __shared__ float red[4][256];
  const int tid = threadIdx.x;
  const int grp = tid >> 6;          // edge within block (4/block)
  const int c4  = tid & 63;          // float4 column chunk
  const int e   = blockIdx.x * 4 + grp;
  float4 v = make_float4(0.f, 0.f, 0.f, 0.f);
  if (e < E) {
    float4 a = ((const float4*)(P1 + (size_t)src[e]  * 256))[c4];
    float4 b = ((const float4*)(P2 + (size_t)dst[e]  * 256))[c4];
    float4 c = ((const float4*)(P3 + (size_t)face[e] * 256))[c4];
    v.x = a.x + b.x + c.x; v.y = a.y + b.y + c.y;
    v.z = a.z + b.z + c.z; v.w = a.w + b.w + c.w;
    ((float4*)(msg + (size_t)e * 256))[c4] = v;
  }
  red[grp][c4*4+0] = v.x; red[grp][c4*4+1] = v.y; red[grp][c4*4+2] = v.z; red[grp][c4*4+3] = v.w;
  __syncthreads();
  atomicAdd(&colSum[tid], red[0][tid] + red[1][tid] + red[2][tid] + red[3][tid]);
  __syncthreads();
  red[grp][c4*4+0] = v.x*v.x; red[grp][c4*4+1] = v.y*v.y;
  red[grp][c4*4+2] = v.z*v.z; red[grp][c4*4+3] = v.w*v.w;
  __syncthreads();
  atomicAdd(&colSumSq[tid], red[0][tid] + red[1][tid] + red[2][tid] + red[3][tid]);
}

// y[dst[e]] += relu(bn(h[e]))   (segment_sum via f32 atomics)
__global__ __launch_bounds__(256) void scatter_kernel(
    const float* __restrict__ h, const int* __restrict__ dst,
    const float* __restrict__ scale, const float* __restrict__ shift,
    float* __restrict__ y, int E)
{
  const int tid = threadIdx.x;
  const int grp = tid >> 5;
  const int c4  = tid & 31;
  const int e   = blockIdx.x * 8 + grp;
  if (e >= E) return;
  const int d = dst[e];
  float4 hv = ((const float4*)(h + (size_t)e * 128))[c4];
  float4 sc = ((const float4*)scale)[c4];
  float4 sh = ((const float4*)shift)[c4];
  float* yr = y + (size_t)d * 128 + c4 * 4;
  atomicAdd(&yr[0], fmaxf(hv.x * sc.x + sh.x, 0.f));
  atomicAdd(&yr[1], fmaxf(hv.y * sc.y + sh.y, 0.f));
  atomicAdd(&yr[2], fmaxf(hv.z * sc.z + sh.z, 0.f));
  atomicAdd(&yr[3], fmaxf(hv.w * sc.w + sh.w, 0.f));
}

__global__ void zero_kernel(float* __restrict__ p, int n) {
  for (int i = blockIdx.x * blockDim.x + threadIdx.x; i < n; i += gridDim.x * blockDim.x)
    p[i] = 0.0f;
}

// scale = g * rsqrt(var + eps), shift = b - mean*scale   (biased batch stats)
__global__ void bn_finalize_kernel(const float* __restrict__ sum, const float* __restrict__ sumsq,
                                   const float* __restrict__ g, const float* __restrict__ b,
                                   float* __restrict__ scale, float* __restrict__ shift,
                                   float invCnt, int C)
{
  int c = threadIdx.x;
  if (c < C) {
    float m = sum[c] * invCnt;
    float v = sumsq[c] * invCnt - m * m;
    float s = g[c] * rsqrtf(v + 1e-5f);
    scale[c] = s;
    shift[c] = b[c] - m * s;
  }
}

extern "C" void kernel_launch(void* const* d_in, const int* in_sizes, int n_in,
                              void* d_out, int out_size, void* d_ws, size_t ws_size,
                              hipStream_t stream)
{
  (void)in_sizes; (void)n_in; (void)out_size; (void)ws_size;
  const int N = 50000, F = 25000, E = 800000, H = 128;
  const int H2 = 2 * H;

  const float* edge_rep = (const float*)d_in[0];
  const float* face_rep = (const float*)d_in[1];
  const int*   eidx     = (const int*)d_in[2];
  const int*   face_idx = (const int*)d_in[3];
  const float* W1  = (const float*)d_in[4];
  const float* W2  = (const float*)d_in[5];
  const float* W3  = (const float*)d_in[6];
  const float* g1a = (const float*)d_in[7];
  const float* b1a = (const float*)d_in[8];
  const float* Wm1 = (const float*)d_in[9];
  const float* g1b = (const float*)d_in[10];
  const float* b1b = (const float*)d_in[11];
  const float* L1  = (const float*)d_in[12];
  const float* g2a = (const float*)d_in[13];
  const float* b2a = (const float*)d_in[14];
  const float* L2  = (const float*)d_in[15];
  const float* g2b = (const float*)d_in[16];
  const float* b2b = (const float*)d_in[17];
  const float* L3  = (const float*)d_in[18];
  const float* bias3 = (const float*)d_in[19];

  const int* src = eidx;          // edge_index[0]
  const int* dst = eidx + E;      // edge_index[1]

  float* ws = (float*)d_ws;
  size_t o = 0;
  float* P1   = ws + o; o += (size_t)N * H2;
  float* P2   = ws + o; o += (size_t)N * H2;
  float* P3   = ws + o; o += (size_t)F * H2;
  float* msg  = ws + o; o += (size_t)E * H2;
  float* hbuf = ws + o; o += (size_t)E * H;
  float* ybuf = ws + o; o += (size_t)N * H;
  float* stats = ws + o; o += 4096;
  float* t1 = P1;   // P1/P2 dead after msg phase -> reuse for mlp2 temps
  float* t2 = P2;

  float* sum1a = stats + 0,    *sq1a = stats + 256,  *sc1a = stats + 512,  *sh1a = stats + 768;
  float* sum1b = stats + 1024, *sq1b = stats + 1152, *sc1b = stats + 1280, *sh1b = stats + 1408;
  float* sum2a = stats + 1536, *sq2a = stats + 1792, *sc2a = stats + 2048, *sh2a = stats + 2304;
  float* sum2b = stats + 2560, *sq2b = stats + 2816, *sc2b = stats + 3072, *sh2b = stats + 3328;

  zero_kernel<<<32, 256, 0, stream>>>(stats, 4096);
  zero_kernel<<<2048, 256, 0, stream>>>(ybuf, N * H);

  auto gx = [](int M) { return (M + BM - 1) / BM; };
  dim3 blk(256);

  // projections (N=256 -> grid.y = 2)
  gemm_bf16_wmma<<<dim3(gx(N), H2 / BN), blk, 0, stream>>>(edge_rep, N, H, W1, H2, P1, H2,
      nullptr, nullptr, nullptr, nullptr, nullptr);
  gemm_bf16_wmma<<<dim3(gx(N), H2 / BN), blk, 0, stream>>>(edge_rep, N, H, W2, H2, P2, H2,
      nullptr, nullptr, nullptr, nullptr, nullptr);
  gemm_bf16_wmma<<<dim3(gx(F), H2 / BN), blk, 0, stream>>>(face_rep, F, H, W3, H2, P3, H2,
      nullptr, nullptr, nullptr, nullptr, nullptr);

  // gather + BN1a stats
  msg_kernel<<<(E + 3) / 4, blk, 0, stream>>>(P1, P2, P3, src, dst, face_idx, msg, sum1a, sq1a, E);
  bn_finalize_kernel<<<1, 256, 0, stream>>>(sum1a, sq1a, g1a, b1a, sc1a, sh1a, 1.0f / E, H2);

  // h = relu(bn(msg)) @ Wm1  (N=128 -> grid.y = 1), BN1b stats in epilogue
  gemm_bf16_wmma<<<dim3(gx(E), H / BN), blk, 0, stream>>>(msg, E, H2, Wm1, H, hbuf, H,
      sc1a, sh1a, nullptr, sum1b, sq1b);
  bn_finalize_kernel<<<1, 256, 0, stream>>>(sum1b, sq1b, g1b, b1b, sc1b, sh1b, 1.0f / E, H);

  // y = segment_sum(relu(bn(h)), dst)
  scatter_kernel<<<(E + 7) / 8, blk, 0, stream>>>(hbuf, dst, sc1b, sh1b, ybuf, E);

  // mlp2
  gemm_bf16_wmma<<<dim3(gx(N), H2 / BN), blk, 0, stream>>>(ybuf, N, H, L1, H2, t1, H2,
      nullptr, nullptr, nullptr, sum2a, sq2a);
  bn_finalize_kernel<<<1, 256, 0, stream>>>(sum2a, sq2a, g2a, b2a, sc2a, sh2a, 1.0f / N, H2);
  gemm_bf16_wmma<<<dim3(gx(N), H2 / BN), blk, 0, stream>>>(t1, N, H2, L2, H2, t2, H2,
      sc2a, sh2a, nullptr, sum2b, sq2b);
  bn_finalize_kernel<<<1, 256, 0, stream>>>(sum2b, sq2b, g2b, b2b, sc2b, sh2b, 1.0f / N, H2);
  gemm_bf16_wmma<<<dim3(gx(N), H / BN), blk, 0, stream>>>(t2, N, H2, L3, H, (float*)d_out, H,
      sc2b, sh2b, bias3, nullptr, nullptr);
}
